// SkipGramNegativeSampleModel_74491912782048
// MI455X (gfx1250) — matrix operand order repaired and checked by reading, not compiled, not run on hardware
//
#include <hip/hip_runtime.h>
#include <hip/hip_bf16.h>

// SkipGram negative-sampling loss on MI455X (gfx1250, wave32, WMMA).
//
// Strategy (memory-bound gather, ~200MB HBM @ 23.3TB/s ~= 9us):
//  - 1 block per batch row b (1024 blocks, 256 threads = 8 wave32s).
//  - Stage embed[walk[b, 0:80]] (all anchors+positives) and the 304 negative
//    rows into LDS as f16 (~105KB, fits CDNA5's 320KB/WG LDS).
//  - Dot products via v_wmma_f32_16x16x32_f16 tiles (K=128 -> 4 k-steps),
//    harvesting only the valid (anchor, context) entries of each 16x16 tile.
//  - Wave-uniform tile params forced into SGPRs via readfirstlane so the
//    per-tile selection lowers to scalar branches (no exec-mask churn).
//  - Branchless harvest: mask * softplus via cndmask+fma.
//  - Deterministic reduction: LDS tree per block -> partials in d_ws ->
//    single fixed-order reduce kernel.

typedef __attribute__((ext_vector_type(16))) _Float16 v16h;
typedef __attribute__((ext_vector_type(8)))  float    v8f;

#define BATCH   1024
#define WLEN    80      // walk length L
#define NA      76      // anchors per row (A = L - w + 1, w = 5)
#define NNEG    4       // negatives per anchor
#define DIM     128     // embedding dim
#define PAD     136     // LDS row stride in f16 elems (even; breaks bank alignment)
#define WALK_ROWS 81    // 80 walk rows + 1 zero pad row (index 80)
#define NEG_BASE  81    // neg rows start here in LDS
#define NEG_ROWS  (NA * NNEG)            // 304
#define TOT_ROWS  (WALK_ROWS + NEG_ROWS) // 385
#define INV_TOTAL (1.0f / (2.0f * BATCH * NA * NNEG)) // 1/622592

__device__ __forceinline__ float softplusf(float x) {
  // stable: max(x,0) + log1p(exp(-|x|))
  float ax = __builtin_fabsf(x);
  return fmaxf(x, 0.0f) + log1pf(__expf(-ax));
}

// A fragment (16x32 f16, M = anchors a0..a0+15, K = 32*kk .. +31).
// ISA layout: lane l holds row M=l&15; VGPR v packs K pair:
//   v<4 : K0 = 2v + 8*(l>>4);  v>=4 : K0 = 16 + 2(v-4) + 8*(l>>4)   (+32*kk)
__device__ __forceinline__ v16h load_A_frag(const unsigned* su, int a0, int kk, int lane) {
  int M  = lane & 15;
  int hi = (lane >> 4) & 1;
  int rbase = (a0 + M) * PAD + 32 * kk;
  union { v16h v; unsigned u[8]; } f;
#pragma unroll
  for (int v = 0; v < 8; ++v) {
    int k0 = (v < 4 ? 2 * v : 16 + 2 * (v - 4)) + hi * 8;
    f.u[v] = su[(rbase + k0) >> 1];
  }
  return f.v;
}

// B fragment (32x16 f16, column N = lane&15 is one context row in LDS).
// ISA layout: VGPR v packs K pair: v<4 : K0 = 2v; v>=4 : K0 = 16 + 2(v-4) (+8 for lanes 16..31), col N.
__device__ __forceinline__ v16h load_B_frag(const unsigned* su, int colElemBase, int kk, int lane) {
  int kh = ((lane >> 4) & 1) * 8 + 32 * kk;
  union { v16h v; unsigned u[8]; } f;
#pragma unroll
  for (int v = 0; v < 8; ++v) {
    f.u[v] = su[(colElemBase + kh + (v < 4 ? 2 * v : 16 + 2 * (v - 4))) >> 1];
  }
  return f.v;
}

extern "C" __global__ __launch_bounds__(256)
void sgns_loss_kernel(const int* __restrict__ walk,
                      const int* __restrict__ neg,
                      const float* __restrict__ embed,
                      float* __restrict__ partial) {
  extern __shared__ _Float16 sh[];          // [TOT_ROWS][PAD] f16
  __shared__ float red[256];

  const int b    = blockIdx.x;
  const int tid  = threadIdx.x;
  const int lane = tid & 31;
  // tid>>5 is identical across a wave32; readfirstlane pins it to an SGPR so
  // all tile-selection control flow below is scalar (s_cbranch_scc, no exec churn).
  const int wave = __builtin_amdgcn_readfirstlane(tid >> 5);

  // ---- Stage walk rows (anchors + positives) to LDS as f16; row 80 = zeros.
  const int* wrow = walk + (size_t)b * WLEN;
  for (int idx = tid; idx < WALK_ROWS * DIM; idx += 256) {
    int r = idx >> 7, c = idx & (DIM - 1);
    float v = 0.0f;
    if (r < WLEN) v = embed[(size_t)wrow[r] * DIM + c];
    sh[r * PAD + c] = (_Float16)v;
  }
  // ---- Stage negative rows.
  const int* nrow = neg + (size_t)b * NEG_ROWS;
  for (int idx = tid; idx < NEG_ROWS * DIM; idx += 256) {
    int r = idx >> 7, c = idx & (DIM - 1);
    sh[(NEG_BASE + r) * PAD + c] = (_Float16)embed[(size_t)nrow[r] * DIM + c];
  }
  __syncthreads();

  const unsigned* su = (const unsigned*)sh;
  float acc = 0.0f;

  // 5 anchor groups (a0 = 0,16,32,48,64) x 6 tiles each:
  //   sub 0: pos tile A  (cols = walk rows a0+1 .. a0+16, diag n-m in 0..3)
  //   sub 1: pos tile B  (cols = walk rows a0+17.., catches m+c >= 16)
  //   sub 2..5: neg tile t (cols = 16 neg rows of anchors a0+4t .. a0+4t+3)
  // 30 tiles round-robined over 8 waves; EXEC uniform around WMMA.
  for (int T = wave; T < 30; T += 8) {
    const int g   = __builtin_amdgcn_readfirstlane(T / 6);      // scalar
    const int sub = __builtin_amdgcn_readfirstlane(T - 6 * g);  // scalar
    const int a0  = g * 16;                                     // scalar
    const int N   = lane & 15;

    int colBase;
    if (sub == 0) {                                      // scalar branch
      colBase = (a0 + 1 + N) * PAD;                      // max row 80 (pad)
    } else if (sub == 1) {                               // scalar branch
      int r = a0 + 17 + N; if (r > 80) r = 80;           // clamp into pad row
      colBase = r * PAD;
    } else {
      int t = sub - 2;
      int a = a0 + 4 * t + (N >> 2); if (a > NA - 1) a = NA - 1;
      colBase = (NEG_BASE + a * NNEG + (N & 3)) * PAD;
    }

    v8f c = {};
#pragma unroll
    for (int kk = 0; kk < 4; ++kk) {
      v16h af = load_A_frag(su, a0, kk, lane);
      v16h bf = load_B_frag(su, colBase, kk, lane);
      c = __builtin_amdgcn_wmma_f32_16x16x32_f16(
          /*neg_a=*/false, af, /*neg_b=*/false, bf,
          /*c_mod=*/(short)0, c, /*reuse_a=*/false, /*reuse_b=*/false);
    }

    // Branchless harvest. C/D layout: VGPR r -> M = r + 8*(lane>>4), N = lane&15.
    // Pos terms contribute softplus(-d), neg terms softplus(+d); sign is
    // scalar per tile. Validity becomes a 0/1 weight (v_cndmask + fma).
    const int   mh  = ((lane >> 4) & 1) * 8;
    const float sgn = (sub <= 1) ? -1.0f : 1.0f;   // scalar
#pragma unroll
    for (int r = 0; r < 8; ++r) {
      int m = r + mh;
      int a = a0 + m;
      int valid;
      if (sub == 0) {                                // scalar branch
        int off = N - m;                             // c index 0..3, m+c <= 15
        valid = (off >= 0) & (off <= 3);
      } else if (sub == 1) {                         // scalar branch
        int off = N + 16 - m;                        // c index 0..3, m+c >= 16
        valid = (off >= 0) & (off <= 3);
      } else {
        int t = sub - 2;
        valid = (m == 4 * t + (N >> 2));
      }
      valid &= (a < NA);
      float w = valid ? 1.0f : 0.0f;                 // v_cndmask, no divergence
      acc = fmaf(w, softplusf(sgn * c[r]), acc);
    }
  }

  // ---- Deterministic block reduction.
  red[tid] = acc;
  __syncthreads();
#pragma unroll
  for (int s = 128; s > 0; s >>= 1) {
    if (tid < s) red[tid] += red[tid + s];
    __syncthreads();
  }
  if (tid == 0) partial[b] = red[0];
}

extern "C" __global__ __launch_bounds__(256)
void sgns_reduce_kernel(const float* __restrict__ partial, float* __restrict__ out) {
  __shared__ float red[256];
  int tid = threadIdx.x;
  float s = partial[tid] + partial[tid + 256] + partial[tid + 512] + partial[tid + 768];
  red[tid] = s;
  __syncthreads();
#pragma unroll
  for (int st = 128; st > 0; st >>= 1) {
    if (tid < st) red[tid] += red[tid + st];
    __syncthreads();
  }
  if (tid == 0) *out = red[0] * INV_TOTAL;
}

extern "C" void kernel_launch(void* const* d_in, const int* in_sizes, int n_in,
                              void* d_out, int out_size, void* d_ws, size_t ws_size,
                              hipStream_t stream) {
  (void)in_sizes; (void)n_in; (void)out_size; (void)ws_size;
  const int*   walk  = (const int*)d_in[0];   // (1024, 80) int32
  const int*   neg   = (const int*)d_in[1];   // (1024, 76, 4) int32
  const float* embed = (const float*)d_in[2]; // (1e6, 128) f32
  float* out     = (float*)d_out;             // scalar loss
  float* partial = (float*)d_ws;              // 1024 floats of scratch (4KB)

  size_t shmem = (size_t)TOT_ROWS * PAD * 2;  // ~105KB f16, OK on CDNA5 (320KB/WG)
  sgns_loss_kernel<<<BATCH, 256, shmem, stream>>>(walk, neg, embed, partial);
  sgns_reduce_kernel<<<1, 256, 0, stream>>>(partial, out);
}
